// Attention_3049426780465
// MI455X (gfx1250) — matrix operand order repaired and checked by reading, not compiled, not run on hardware
//
#include <hip/hip_runtime.h>
#include <hip/hip_bf16.h>
#include <math.h>

#define B_  2
#define N_  4096
#define C_  512
#define H_  8
#define HD_ 64

typedef _Float16 v8h  __attribute__((ext_vector_type(8)));
typedef _Float16 v16h __attribute__((ext_vector_type(16)));
typedef float    v8f  __attribute__((ext_vector_type(8)));

static __device__ __forceinline__ v8f wmma_f16(v16h a, v16h b, v8f c) {
  // D = A(16x32 f16) * B(32x16 f16) + C(16x16 f32)
  return __builtin_amdgcn_wmma_f32_16x16x32_f16(false, a, false, b, (short)0, c,
                                                false, false);
}

// A-operand (16x32 f16) from row-major matrix, tile origin `base`, leading dim ld.
// Contiguous k order used here: lanes 0-15 hold K 0..15 of row (lane&15);
//                 lanes 16-31 hold K 16..31.
static __device__ __forceinline__ v16h load_a(const _Float16* __restrict__ base,
                                              int ld, int lane) {
  const int r  = lane & 15;
  const int hi = lane >> 4;
  const _Float16* p = base + (size_t)r * ld + hi * 16;
  v8h x0 = *(const v8h*)(p);
  v8h x1 = *(const v8h*)(p + 8);
  return __builtin_shufflevector(x0, x1, 0,1,2,3,4,5,6,7,8,9,10,11,12,13,14,15);
}

// B-operand (32x16 f16) whose columns are rows of a row-major matrix (NT GEMM).
// Contiguous k order used here: lanes 0-15 hold K 0..15 of column (lane&15);
//                 lanes 16-31 hold K 16..31.
static __device__ __forceinline__ v16h load_b(const _Float16* __restrict__ base,
                                              int ld, int lane) {
  const int r  = lane & 15;
  const int hi = lane >> 4;
  const _Float16* p = base + (size_t)r * ld + hi * 16;
  v8h x0 = *(const v8h*)(p);
  v8h x1 = *(const v8h*)(p + 8);
  return __builtin_shufflevector(x0, x1, 0,1,2,3,4,5,6,7,8,9,10,11,12,13,14,15);
}

// ---------------------------------------------------------------- conversions
__global__ __launch_bounds__(256) void attn_cvt_f32_f16(
    const float* __restrict__ s, _Float16* __restrict__ d, int n) {
  int i = blockIdx.x * 256 + threadIdx.x;
  if (i < n) d[i] = (_Float16)s[i];
}

// ------------------------------------------------------------------ QKV GEMM
// out[m,d] = sum_c x[m,c] * w_qkv[d,c]; scatter into Q/K (b,h,n,hd) and
// V transposed (b,h,hd,n), all f16. Wave tile: 32(M) x 64(D), 8 accumulators.
// Block: 8 waves covering 64(M) x 256(D).
__global__ __launch_bounds__(256) void attn_qkv_gemm(
    const _Float16* __restrict__ xh,   // [B*N, C]
    const _Float16* __restrict__ wh,   // [3C, C]
    _Float16* __restrict__ Qh,         // [B,H,N,HD]
    _Float16* __restrict__ Kh,         // [B,H,N,HD]
    _Float16* __restrict__ Vt) {       // [B,H,HD,N]
  const int lane = threadIdx.x & 31;
  const int wave = threadIdx.x >> 5;
  const int m0 = blockIdx.x * 64 + (wave >> 2) * 32;
  const int d0 = blockIdx.y * 256 + (wave & 3) * 64;

  v8f acc[8] = {};
  const _Float16* ap = xh + (size_t)m0 * C_;
  const _Float16* bp = wh + (size_t)d0 * C_;
#pragma unroll 2
  for (int k = 0; k < C_; k += 32) {
    // All operands in distinct locals: one load clause, then 8 back-to-back WMMAs.
    v16h a0 = load_a(ap + k,            C_, lane);
    v16h a1 = load_a(ap + 16 * C_ + k,  C_, lane);
    v16h b0 = load_b(bp + k,            C_, lane);
    v16h b1 = load_b(bp + 16 * C_ + k,  C_, lane);
    v16h b2 = load_b(bp + 32 * C_ + k,  C_, lane);
    v16h b3 = load_b(bp + 48 * C_ + k,  C_, lane);
    acc[0] = wmma_f16(a0, b0, acc[0]);
    acc[1] = wmma_f16(a0, b1, acc[1]);
    acc[2] = wmma_f16(a0, b2, acc[2]);
    acc[3] = wmma_f16(a0, b3, acc[3]);
    acc[4] = wmma_f16(a1, b0, acc[4]);
    acc[5] = wmma_f16(a1, b1, acc[5]);
    acc[6] = wmma_f16(a1, b2, acc[6]);
    acc[7] = wmma_f16(a1, b3, acc[7]);
  }

  // d0 is 64-aligned -> (t,h) constant per wave; e = local offset in head dim.
  const int t   = d0 >> 9;
  const int h   = (d0 >> 6) & (H_ - 1);
  const int col = lane & 15;
  const int hi  = lane >> 4;
#pragma unroll
  for (int i = 0; i < 2; ++i) {
#pragma unroll
    for (int j = 0; j < 4; ++j) {
#pragma unroll
      for (int r = 0; r < 8; ++r) {
        const int m = m0 + i * 16 + hi * 8 + r;
        const int b = m >> 12;            // m / N_
        const int n = m & (N_ - 1);
        const int e = j * 16 + col;
        const _Float16 v = (_Float16)acc[i * 4 + j][r];
        const size_t bh = (size_t)b * H_ + h;
        if (t == 0)      Qh[(bh * N_ + n) * HD_ + e] = v;
        else if (t == 1) Kh[(bh * N_ + n) * HD_ + e] = v;
        else             Vt[(bh * HD_ + e) * N_ + n] = v;
      }
    }
  }
}

// ------------------------------------------------------------ flash attention
// One wave owns 16 queries for one (b,h); streams all 4096 keys in blocks of
// 32. Computes S^T = K*Q^T so softmaxed probs feed P*V as the B operand with
// only lane-pair shuffles (no LDS transpose). O accumulated as O^T (hd x q).
// Software-pipelined: V tiles are issued before the softmax VALU work, next K tiles
// right after it, so the WMMA pipe never waits on cold loads.
__global__ __launch_bounds__(128) void attn_flash(
    const _Float16* __restrict__ Qh, const _Float16* __restrict__ Kh,
    const _Float16* __restrict__ Vt, _Float16* __restrict__ Oh) {  // [B*N, C]
  const int lane = threadIdx.x & 31;
  const int wave = threadIdx.x >> 5;   // 0..3
  const int bh = blockIdx.y;           // b*H + h
  const int q0 = blockIdx.x * 64 + wave * 16;

  const _Float16* Qp = Qh + (size_t)bh * N_ * HD_;  // [N, HD]
  const _Float16* Kp = Kh + (size_t)bh * N_ * HD_;  // [N, HD]
  const _Float16* Vp = Vt + (size_t)bh * HD_ * N_;  // [HD, N]

  // Q^T as B-operand (K-dim = channel), loaded once per wave.
  const v16h qb0 = load_b(Qp + (size_t)q0 * HD_ + 0,  HD_, lane);
  const v16h qb1 = load_b(Qp + (size_t)q0 * HD_ + 32, HD_, lane);

  v8f o0 = {}, o1 = {}, o2 = {}, o3 = {};          // O^T: 64(hd) x 16(q)
  float m_run = -__builtin_inff();
  float l_run = 0.0f;
  const float c1 = 0.125f * 1.44269504088896340736f;  // hd^-0.5 * log2(e)
  const bool lowhalf = (lane < 16);

  // Preload first key block's K tiles.
  v16h ka0 = load_a(Kp,                 HD_, lane);
  v16h ka1 = load_a(Kp + 32,            HD_, lane);
  v16h ka2 = load_a(Kp + 16 * HD_,      HD_, lane);
  v16h ka3 = load_a(Kp + 16 * HD_ + 32, HD_, lane);

  for (int kb = 0; kb < N_; kb += 32) {
    // S^T tiles: T0 = keys kb..kb+15, T1 = keys kb+16..kb+31 (x 16 queries)
    v8f s0 = {}, s1 = {};
    s0 = wmma_f16(ka0, qb0, s0);
    s0 = wmma_f16(ka1, qb1, s0);
    s1 = wmma_f16(ka2, qb0, s1);
    s1 = wmma_f16(ka3, qb1, s1);

    // Issue V loads for this block early (consumed after softmax).
    const _Float16* vp = Vp + kb;
    v16h va0 = load_a(vp,            N_, lane);
    v16h va1 = load_a(vp + 16 * N_,  N_, lane);
    v16h va2 = load_a(vp + 32 * N_,  N_, lane);
    v16h va3 = load_a(vp + 48 * N_,  N_, lane);

    // Streaming prefetch ~3 key-blocks ahead (K block = 32 rows x 128B,
    // contiguous; 32 lanes x 128B cover it exactly).
    if (kb + 128 <= N_) {
      const char* kpf = (const char*)(Kp + (size_t)(kb + 96) * HD_) + lane * 128;
      __builtin_prefetch(kpf, 0, 3);
      const char* vpf = (const char*)(Vp + (size_t)lane * N_ + (kb + 96));
      __builtin_prefetch(vpf, 0, 3);
      __builtin_prefetch(vpf + (size_t)32 * N_ * 2, 0, 3);
    }

    // per-query online softmax (keys live in VGPRs + partner lane)
    float mt = -__builtin_inff();
#pragma unroll
    for (int r = 0; r < 8; ++r) {
      mt = fmaxf(mt, s0[r]);
      mt = fmaxf(mt, s1[r]);
    }
    float mth = mt * c1;
    mth = fmaxf(mth, __shfl_xor(mth, 16, 32));
    const float m_new = fmaxf(m_run, mth);
    const float corr  = exp2f(m_run - m_new);   // 0 on first iteration

    float p0[8], p1[8];
    float rs = 0.0f;
#pragma unroll
    for (int r = 0; r < 8; ++r) {
      p0[r] = exp2f(s0[r] * c1 - m_new);
      p1[r] = exp2f(s1[r] * c1 - m_new);
      rs += p0[r] + p1[r];
    }
    l_run = l_run * corr + rs;   // per-lane half; partner half folded at end
    m_run = m_new;

    // Load next key block's K tiles while the VALU work above is in flight.
    if (kb + 32 < N_) {
      const _Float16* kn = Kp + (size_t)(kb + 32) * HD_;
      ka0 = load_a(kn,                 HD_, lane);
      ka1 = load_a(kn + 32,            HD_, lane);
      ka2 = load_a(kn + 16 * HD_,      HD_, lane);
      ka3 = load_a(kn + 16 * HD_ + 32, HD_, lane);
    }

#pragma unroll
    for (int r = 0; r < 8; ++r) {
      o0[r] *= corr; o1[r] *= corr; o2[r] *= corr; o3[r] *= corr;
    }

    // Build P as B-operand (32 keys x 16 queries). C-layout key rows 0-7/8-15
    // sit on opposite lane halves vs this code's B order K 0-15/16-31 -> one xor-16 swap.
    v16h pb;
#pragma unroll
    for (int r = 0; r < 8; ++r) {
      const float send = lowhalf ? p1[r] : p0[r];
      const float recv = __shfl_xor(send, 16, 32);
      pb[r]     = (_Float16)(lowhalf ? p0[r] : recv);  // K 0..7  / 16..23
      pb[r + 8] = (_Float16)(lowhalf ? recv  : p1[r]); // K 8..15 / 24..31
    }

    // O^T += Vt * P  (A = Vt rows = hd, K = keys)
    o0 = wmma_f16(va0, pb, o0);
    o1 = wmma_f16(va1, pb, o1);
    o2 = wmma_f16(va2, pb, o2);
    o3 = wmma_f16(va3, pb, o3);
  }

  const float l_tot = l_run + __shfl_xor(l_run, 16, 32);
  const float inv = 1.0f / l_tot;
  const int b = bh >> 3, h = bh & (H_ - 1);
  const int col = lane & 15, hi = lane >> 4;
  _Float16* orow = Oh + ((size_t)b * N_ + (q0 + col)) * C_ + h * HD_;
  const v8f oo[4] = {o0, o1, o2, o3};
#pragma unroll
  for (int t = 0; t < 4; ++t) {
#pragma unroll
    for (int r = 0; r < 8; ++r) {
      orow[t * 16 + hi * 8 + r] = (_Float16)(oo[t][r] * inv);
    }
  }
}

// ------------------------------------------------------------------ proj GEMM
// out[m,d] = sum_c O[m,c] * w_proj[d,c] + b_proj[d], f32 output.
// Wave tile: 32(M) x 64(D), 8 accumulators.
__global__ __launch_bounds__(256) void attn_proj_gemm(
    const _Float16* __restrict__ Oh,   // [B*N, C]
    const _Float16* __restrict__ wh,   // [C, C]
    const float* __restrict__ bias,    // [C]
    float* __restrict__ out) {         // [B*N, C]
  const int lane = threadIdx.x & 31;
  const int wave = threadIdx.x >> 5;
  const int m0 = blockIdx.x * 64 + (wave >> 2) * 32;
  const int d0 = blockIdx.y * 256 + (wave & 3) * 64;

  v8f acc[8] = {};
  const _Float16* ap = Oh + (size_t)m0 * C_;
  const _Float16* bp = wh + (size_t)d0 * C_;
#pragma unroll 2
  for (int k = 0; k < C_; k += 32) {
    v16h a0 = load_a(ap + k,            C_, lane);
    v16h a1 = load_a(ap + 16 * C_ + k,  C_, lane);
    v16h b0 = load_b(bp + k,            C_, lane);
    v16h b1 = load_b(bp + 16 * C_ + k,  C_, lane);
    v16h b2 = load_b(bp + 32 * C_ + k,  C_, lane);
    v16h b3 = load_b(bp + 48 * C_ + k,  C_, lane);
    acc[0] = wmma_f16(a0, b0, acc[0]);
    acc[1] = wmma_f16(a0, b1, acc[1]);
    acc[2] = wmma_f16(a0, b2, acc[2]);
    acc[3] = wmma_f16(a0, b3, acc[3]);
    acc[4] = wmma_f16(a1, b0, acc[4]);
    acc[5] = wmma_f16(a1, b1, acc[5]);
    acc[6] = wmma_f16(a1, b2, acc[6]);
    acc[7] = wmma_f16(a1, b3, acc[7]);
  }

  const int col = lane & 15;
  const int hi  = lane >> 4;
#pragma unroll
  for (int i = 0; i < 2; ++i) {
#pragma unroll
    for (int j = 0; j < 4; ++j) {
      const int d = d0 + j * 16 + col;
      const float bj = bias[d];
#pragma unroll
      for (int r = 0; r < 8; ++r) {
        const int m = m0 + i * 16 + hi * 8 + r;
        out[(size_t)m * C_ + d] = acc[i * 4 + j][r] + bj;
      }
    }
  }
}

// ----------------------------------------------------------------- launcher
extern "C" void kernel_launch(void* const* d_in, const int* in_sizes, int n_in,
                              void* d_out, int out_size, void* d_ws,
                              size_t ws_size, hipStream_t stream) {
  const float* x      = (const float*)d_in[0];
  const float* w_qkv  = (const float*)d_in[1];
  const float* w_proj = (const float*)d_in[2];
  const float* b_proj = (const float*)d_in[3];

  char* ws = (char*)d_ws;
  size_t off = 0;
  _Float16* xh     = (_Float16*)(ws + off); off += (size_t)B_ * N_ * C_ * 2;
  _Float16* wqkvh  = (_Float16*)(ws + off); off += (size_t)3 * C_ * C_ * 2;
  _Float16* wprojh = (_Float16*)(ws + off); off += (size_t)C_ * C_ * 2;
  _Float16* Qh     = (_Float16*)(ws + off); off += (size_t)B_ * H_ * N_ * HD_ * 2;
  _Float16* Kh     = (_Float16*)(ws + off); off += (size_t)B_ * H_ * N_ * HD_ * 2;
  _Float16* Vt     = (_Float16*)(ws + off); off += (size_t)B_ * H_ * N_ * HD_ * 2;
  _Float16* Oh     = (_Float16*)(ws + off); off += (size_t)B_ * N_ * C_ * 2;
  (void)ws_size; (void)in_sizes; (void)n_in; (void)out_size;

  attn_cvt_f32_f16<<<(B_ * N_ * C_) / 256, 256, 0, stream>>>(x, xh, B_ * N_ * C_);
  attn_cvt_f32_f16<<<(3 * C_ * C_) / 256, 256, 0, stream>>>(w_qkv, wqkvh, 3 * C_ * C_);
  attn_cvt_f32_f16<<<(C_ * C_) / 256, 256, 0, stream>>>(w_proj, wprojh, C_ * C_);

  attn_qkv_gemm<<<dim3((B_ * N_) / 64, (3 * C_) / 256), 256, 0, stream>>>(
      xh, wqkvh, Qh, Kh, Vt);

  attn_flash<<<dim3(N_ / 64, B_ * H_), 128, 0, stream>>>(Qh, Kh, Vt, Oh);

  attn_proj_gemm<<<dim3((B_ * N_) / 64, C_ / 256), 256, 0, stream>>>(
      Oh, wprojh, b_proj, (float*)d_out);
}
